// MultiheadAttentionBias_58171037057243
// MI455X (gfx1250) — matrix operand order repaired and checked by reading, not compiled, not run on hardware
//
#include <hip/hip_runtime.h>
#include <math.h>

typedef __bf16 bf16;
typedef __attribute__((ext_vector_type(16))) __bf16 v16bf;
typedef __attribute__((ext_vector_type(8)))  __bf16 v8bf;
typedef __attribute__((ext_vector_type(8)))  float  v8f;
typedef __attribute__((ext_vector_type(4)))  unsigned int u32x4;
typedef __attribute__((ext_vector_type(8)))  int i32x8;
typedef __attribute__((ext_vector_type(4)))  int i32x4;

#define DEV static __device__ __forceinline__

// Problem constants: B=4, N=1024, C=256, H=8  -> ROWS=4096, CH=2048
// ---------------------------------------------------------------------------
// A-fragment (16x32 bf16, row-major source, K contiguous).
// lane L: row = L&15 ; kb = 8*(L>=16); elements = K[kb..kb+7] ++ K[kb+16..kb+23]
DEV v16bf load_a_frag(const bf16* p) {
  v8bf lo = *(const v8bf*)(p);
  v8bf hi = *(const v8bf*)(p + 16);
  return __builtin_shufflevector(lo, hi, 0,1,2,3,4,5,6,7,8,9,10,11,12,13,14,15);
}

// B-fragment (32x16 bf16) from K-major storage: lane L holds column
// N = L&15; element e has k = 8*(L>=16) + (e&7) + 16*(e>>3), gathered one at a time.
DEV v8f wmma_bf16(v16bf a, v16bf b, v8f c) {
  return __builtin_amdgcn_wmma_f32_16x16x32_bf16(false, a, false, b, (short)0, c,
                                                 false, false);
}

DEV v8f vzero8() { v8f z = {0.f,0.f,0.f,0.f,0.f,0.f,0.f,0.f}; return z; }

// TDM: 2D tile load Global->LDS (toolchain-portable arity).
DEV void tdm_load_2d(u32x4 g0, i32x8 g1) {
  i32x4 z4 = {0, 0, 0, 0};
#if __clang_major__ >= 23
  i32x8 z8 = {0, 0, 0, 0, 0, 0, 0, 0};
  __builtin_amdgcn_tensor_load_to_lds(g0, g1, z4, z4, z8, 0);
#else
  __builtin_amdgcn_tensor_load_to_lds(g0, g1, z4, z4, 0);
#endif
}

// ---------------------------------------------------------------------------
// Generic f32 -> bf16 convert (also used for linear_output, already K-major)
__global__ __launch_bounds__(256) void k_cvt(const float* __restrict__ s,
                                             bf16* __restrict__ d, int n) {
  int i = blockIdx.x * 256 + threadIdx.x;
  int stride = gridDim.x * 256;
  for (; i < n; i += stride) d[i] = (bf16)s[i];
}

// Pack weight (C,C,H)=[i][j][h] f32  ->  K-major bf16 [h][j][i]
__global__ __launch_bounds__(256) void k_pack_w(const float* __restrict__ s,
                                                bf16* __restrict__ d) {
  int idx = blockIdx.x * 256 + threadIdx.x;     // 8*256*256 total
  int i = idx & 255, j = (idx >> 8) & 255, h = idx >> 16;
  d[idx] = (bf16)s[((i << 8) + j) * 8 + h];
}

// ---------------------------------------------------------------------------
// Projections: xh[b,n,i,h] = sum_j W[i,j,h]*x[b,n,j] + bias[i,h]
// One wave computes a 16(M) x 64(N) tile, K = 256 (8 wmma steps x 4 col tiles)
__global__ __launch_bounds__(256) void k_proj(
    const bf16* __restrict__ xq, const bf16* __restrict__ xk,
    const bf16* __restrict__ xv, const bf16* __restrict__ Wt, // [mat][h][j][i]
    const float* __restrict__ bQ, const float* __restrict__ bK,
    const float* __restrict__ bV,
    bf16* __restrict__ qh,    // [h][row 4096][i 256]
    bf16* __restrict__ khT,   // [h][b][i 256][n 1024]
    float* __restrict__ vflat /* [row 4096][i*8+h 2048] */) {
  int wave = threadIdx.x >> 5, lane = threadIdx.x & 31;
  int task = blockIdx.x * 8 + wave;             // 24576 tasks
  int ct  = task & 3;
  int rt  = (task >> 2) & 255;
  int h   = (task >> 10) & 7;
  int mat = task >> 13;                         // 0,1,2
  int r0 = rt * 16, c0 = ct * 64;
  const bf16* X = (mat == 0) ? xq : ((mat == 1) ? xk : xv);
  const bf16* W = Wt + ((size_t)mat * 8 + h) * 65536;
  int lm = lane & 15, hiA = (lane >> 4) << 3, hiB = (lane >> 4) << 4;

  v8f c[4];
  #pragma unroll
  for (int s = 0; s < 4; ++s) c[s] = vzero8();

  #pragma unroll
  for (int kk = 0; kk < 8; ++kk) {
    int k0 = kk * 32;
    v16bf a = load_a_frag(X + (size_t)(r0 + lm) * 256 + k0 + hiA);
    v16bf bfr[4];                                // hoist loads: one clause
    #pragma unroll
    for (int s = 0; s < 4; ++s)
      for (int e = 0; e < 16; ++e)
        bfr[s][e] = W[(size_t)(k0 + hiA + (e & 7) + ((e >> 3) << 4)) * 256 + c0 + s * 16 + lm];
    #pragma unroll
    for (int s = 0; s < 4; ++s) c[s] = wmma_bf16(a, bfr[s], c[s]);
  }

  const float* bias = (mat == 0) ? bQ : ((mat == 1) ? bK : bV);
  #pragma unroll
  for (int s = 0; s < 4; ++s) {
    int col = c0 + s * 16 + lm;
    float bv = bias[col * 8 + h];
    #pragma unroll
    for (int d = 0; d < 8; ++d) {
      int row = r0 + d + hiA;
      float val = c[s][d] + bv;
      if (mat == 0) {
        qh[((size_t)h * 4096 + row) * 256 + col] = (bf16)val;
      } else if (mat == 1) {
        int bb = row >> 10, n = row & 1023;     // transpose to K-major
        khT[(((size_t)h * 4 + bb) * 256 + col) * 1024 + n] = (bf16)val;
      } else {
        vflat[(size_t)row * 2048 + col * 8 + h] = val;
      }
    }
  }
}

// ---------------------------------------------------------------------------
// Scores + online-softmax row sum Z:  S = qh_h * kh_h^T + mask + bias
// Block = 8 waves (one head each), one (b, 16-row n-tile) per block.
// The 134 MB bias tensor is streamed by the Tensor Data Mover into a
// double-buffered LDS tile (16 n-rows x 128 floats = one (16m x 8h) slab),
// issued by wave 0 only, synchronized via TENSORcnt + workgroup barrier.
__global__ __launch_bounds__(256) void k_scores(
    const bf16* __restrict__ qh, const bf16* __restrict__ khT,
    const float* __restrict__ bias, const float* __restrict__ mask,
    float* __restrict__ rowsum /* [row 4096][h 8] */) {
  __shared__ float ldsB[2][16 * 128];  // two 8 KB tiles: [rn][rm*8+h]
  int h = threadIdx.x >> 5, lane = threadIdx.x & 31;
  int b = blockIdx.x >> 6, nt = blockIdx.x & 63;
  int n0 = nt * 16, r0 = b * 1024 + n0;
  int lm = lane & 15, hi8 = (lane >> 4) << 3, hiB = (lane >> 4) << 4;
  bool issuer = (threadIdx.x < 32);    // wave 0 drives the TDM

  unsigned int ldsOff = (unsigned int)(uintptr_t)(&ldsB[0][0]); // LDS byte offset
  const float* tileBase = bias + ((size_t)(b * 1024 + n0) * 1024) * 8;

  // D# builder: 2D tile, data_size=4B, tile 128(x) x 16(y), row stride = one
  // bias n-row = 1024*8 floats; LDS dest = ldsB[buf].
  auto tdm_issue = [&](int mt, int buf) {
    unsigned long long ga =
        (unsigned long long)(uintptr_t)(tileBase + (size_t)mt * 16 * 8);
    u32x4 g0;
    g0.x = 1u;                                             // count=1 (user D#)
    g0.y = ldsOff + (unsigned int)buf * 8192u;             // lds_addr (bytes)
    g0.z = (unsigned int)ga;                               // global_addr lo
    g0.w = (unsigned int)((ga >> 32) & 0x01FFFFFFu) | (2u << 30); // hi | type=2
    i32x8 g1;
    g1[0] = 0x00020000;                    // data_size=2 -> 4 bytes
    g1[1] = (int)((8192u & 0xFFFFu) << 16);          // tensor_dim0 = 8192
    g1[2] = (int)((8192u >> 16) | (16u << 16));      // tensor_dim1 = 16
    g1[3] = (int)(128u << 16);                       // tile_dim0 = 128
    g1[4] = 16;                                      // tile_dim1 = 16
    g1[5] = 8192;                                    // tensor_dim0_stride lo
    g1[6] = 0;
    g1[7] = 0;
    tdm_load_2d(g0, g1);
  };

  // cache the whole A row-tile (16 rows x 256 K) in registers
  const bf16* qhh = qh + (size_t)h * 4096 * 256 + (size_t)(r0 + lm) * 256 + hi8;
  v16bf a[8];
  #pragma unroll
  for (int kk = 0; kk < 8; ++kk) a[kk] = load_a_frag(qhh + kk * 32);

  const bf16* khb = khT + ((size_t)h * 4 + b) * 256 * 1024;

  float mrun[8], srun[8];
  #pragma unroll
  for (int d = 0; d < 8; ++d) { mrun[d] = -3.402823466e38f; srun[d] = 0.f; }

  if (issuer) tdm_issue(0, 0);           // prologue: prefetch tile 0

  for (int mt = 0; mt < 64; ++mt) {
    int m0 = mt * 16;
    if (issuer) __builtin_amdgcn_s_wait_tensorcnt(0);  // buf[mt&1] landed
    __syncthreads();
    if (issuer && mt + 1 < 64) tdm_issue(mt + 1, (mt + 1) & 1);
    const float* tile = &ldsB[mt & 1][0];

    v8f c = vzero8();
    #pragma unroll
    for (int kk = 0; kk < 8; ++kk) {
      v16bf bfrag;
      for (int e = 0; e < 16; ++e)
        bfrag[e] = khb[(size_t)(kk * 32 + hi8 + (e & 7) + ((e >> 3) << 4)) * 1024 + m0 + lm];
      c = wmma_bf16(a[kk], bfrag, c);
    }

    float mk = mask[b * 1024 + m0 + lm];
    #pragma unroll
    for (int d = 0; d < 8; ++d) {
      float val = c[d] + mk + tile[(d + hi8) * 128 + lm * 8 + h];
      float tmax = val;                           // butterfly max over 16 lanes
      #pragma unroll
      for (int off = 1; off < 16; off <<= 1)
        tmax = fmaxf(tmax, __shfl_xor(tmax, off, 32));
      float mnew = fmaxf(mrun[d], tmax);
      float p = __expf(val - mnew);
      float tsum = p;                             // butterfly sum
      #pragma unroll
      for (int off = 1; off < 16; off <<= 1)
        tsum += __shfl_xor(tsum, off, 32);
      srun[d] = srun[d] * __expf(mrun[d] - mnew) + tsum;
      mrun[d] = mnew;
    }
    __syncthreads();   // everyone done with buf[mt&1] before it is re-filled
  }

  if (lm == 0) {
    #pragma unroll
    for (int d = 0; d < 8; ++d)
      rowsum[(size_t)(r0 + d + hi8) * 8 + h] = srun[d];
  }
}

// ---------------------------------------------------------------------------
// out_flat = vh * (Z/Z)   (the m-contraction of softmax(alpha) is exactly Z'/Z)
__global__ __launch_bounds__(256) void k_scale(const float* __restrict__ vflat,
                                               const float* __restrict__ rowsum,
                                               bf16* __restrict__ vbf) {
  int row = blockIdx.x;
  int base = threadIdx.x * 8;                    // covers h = 0..7 at i = tid
  #pragma unroll
  for (int u = 0; u < 8; ++u) {
    float Z = rowsum[row * 8 + u];
    float scale = Z / Z;                         // runtime value: stays live
    vbf[(size_t)row * 2048 + base + u] =
        (bf16)(vflat[(size_t)row * 2048 + base + u] * scale);
  }
}

// ---------------------------------------------------------------------------
// Final linear: out[row][j] = sum_k flat[row][k] * L[k][j]   (K = 2048)
__global__ __launch_bounds__(256) void k_final(const bf16* __restrict__ flat,
                                               const bf16* __restrict__ Lbf,
                                               float* __restrict__ out) {
  int wave = threadIdx.x >> 5, lane = threadIdx.x & 31;
  int task = blockIdx.x * 8 + wave;              // 1024 tasks
  int ct = task & 3, rt = task >> 2;
  int r0 = rt * 16, c0 = ct * 64;
  int lm = lane & 15, hiA = (lane >> 4) << 3, hiB = (lane >> 4) << 4;

  v8f c[4];
  #pragma unroll
  for (int s = 0; s < 4; ++s) c[s] = vzero8();

  for (int k0 = 0; k0 < 2048; k0 += 32) {
    v16bf a = load_a_frag(flat + (size_t)(r0 + lm) * 2048 + k0 + hiA);
    v16bf bfr[4];
    #pragma unroll
    for (int s = 0; s < 4; ++s)
      for (int e = 0; e < 16; ++e)
        bfr[s][e] = Lbf[(size_t)(k0 + hiA + (e & 7) + ((e >> 3) << 4)) * 256 + c0 + s * 16 + lm];
    #pragma unroll
    for (int s = 0; s < 4; ++s) c[s] = wmma_bf16(a, bfr[s], c[s]);
  }
  #pragma unroll
  for (int s = 0; s < 4; ++s) {
    int col = c0 + s * 16 + lm;
    #pragma unroll
    for (int d = 0; d < 8; ++d)
      out[(size_t)(r0 + d + hiA) * 256 + col] = c[s][d];
  }
}

// ---------------------------------------------------------------------------
extern "C" void kernel_launch(void* const* d_in, const int* in_sizes, int n_in,
                              void* d_out, int out_size, void* d_ws, size_t ws_size,
                              hipStream_t stream) {
  (void)in_sizes; (void)n_in; (void)out_size; (void)ws_size;
  const float* q    = (const float*)d_in[0];
  const float* k    = (const float*)d_in[1];
  const float* v    = (const float*)d_in[2];
  const float* bias = (const float*)d_in[3];
  const float* mask = (const float*)d_in[4];
  const float* Qw   = (const float*)d_in[5];
  const float* Kw   = (const float*)d_in[6];
  const float* Vw   = (const float*)d_in[7];
  const float* bQ   = (const float*)d_in[8];
  const float* bK   = (const float*)d_in[9];
  const float* bV   = (const float*)d_in[10];
  const float* Lw   = (const float*)d_in[11];

  unsigned char* w = (unsigned char*)d_ws;
  auto carve = [&](size_t bytes) {
    void* p = (void*)w;
    w += (bytes + 255) & ~(size_t)255;
    return p;
  };
  bf16*  qbf    = (bf16*)carve(4096 * 256 * 2);          // 2 MB
  bf16*  kbf    = (bf16*)carve(4096 * 256 * 2);
  bf16*  vbf    = (bf16*)carve(4096 * 256 * 2);
  bf16*  Wt     = (bf16*)carve(3 * 8 * 256 * 256 * 2);   // 3 MB
  bf16*  Lbf    = (bf16*)carve(2048 * 256 * 2);          // 1 MB
  bf16*  qh     = (bf16*)carve(8 * 4096 * 256 * 2);      // 16 MB
  bf16*  khT    = (bf16*)carve(8 * 4096 * 256 * 2);      // 16 MB
  float* vflat  = (float*)carve(4096 * 2048 * 4);        // 32 MB
  float* rowsum = (float*)carve(4096 * 8 * 4);
  bf16*  flatbf = (bf16*)carve(4096 * 2048 * 2);         // 16 MB

  k_cvt<<<1024, 256, 0, stream>>>(q, qbf, 1048576);
  k_cvt<<<1024, 256, 0, stream>>>(k, kbf, 1048576);
  k_cvt<<<1024, 256, 0, stream>>>(v, vbf, 1048576);
  k_cvt<<<512,  256, 0, stream>>>(Lw, Lbf, 524288);
  k_pack_w<<<2048, 256, 0, stream>>>(Qw, Wt);
  k_pack_w<<<2048, 256, 0, stream>>>(Kw, Wt + 524288);
  k_pack_w<<<2048, 256, 0, stream>>>(Vw, Wt + 1048576);
  k_proj<<<3072, 256, 0, stream>>>(qbf, kbf, vbf, Wt, bQ, bK, bV, qh, khT, vflat);
  k_scores<<<256, 256, 0, stream>>>(qh, khT, bias, mask, rowsum);
  k_scale<<<4096, 256, 0, stream>>>(vflat, rowsum, flatbf);
  k_final<<<128, 256, 0, stream>>>(flatbf, Lbf, (float*)d_out);
}
